// LSTM_55001351193085
// MI455X (gfx1250) — matrix operand order repaired and checked by reading, not compiled, not run on hardware
//
#include <hip/hip_runtime.h>

typedef __attribute__((ext_vector_type(16))) __bf16 v16bf;
typedef __attribute__((ext_vector_type(8)))  __bf16 v8bf;
typedef __attribute__((ext_vector_type(8)))  float  v8f;

#define BB 512
#define TT 128
#define DD 512
#define HH 512
#define KK 1024   // D + H
#define NG 2048   // 4*H

__device__ __forceinline__ float fsig(float v) {
    return 1.0f / (1.0f + __expf(-v));
}
__device__ __forceinline__ float ftanh(float v) {
    // stable: e->0 gives -1, e->inf gives +1
    float e = __expf(2.0f * v);
    return 1.0f - 2.0f / (e + 1.0f);
}

// ---------------------------------------------------------------------------
// One-time: W [1024][2048] f32 row-major  ->  WbT [2048][1024] bf16
// (so each WMMA B-tile lane load is 16 contiguous bf16 = 32B)
// ---------------------------------------------------------------------------
__global__ __launch_bounds__(256) void lstm_convW(const float* __restrict__ W,
                                                  __bf16* __restrict__ WbT) {
    __shared__ float tile[32][33];
    const int kb = blockIdx.x * 32;   // 0..1023 in steps of 32
    const int nb = blockIdx.y * 32;   // 0..2047 in steps of 32
    const int lx = threadIdx.x;       // 0..31
    const int ly = threadIdx.y;       // 0..7
#pragma unroll
    for (int i = 0; i < 32; i += 8)
        tile[ly + i][lx] = W[(size_t)(kb + ly + i) * NG + nb + lx];
    __syncthreads();
#pragma unroll
    for (int i = 0; i < 32; i += 8)
        WbT[(size_t)(nb + ly + i) * KK + kb + lx] = (__bf16)tile[lx][ly + i];
}

// ---------------------------------------------------------------------------
// Zero the recurrent state
// ---------------------------------------------------------------------------
__global__ __launch_bounds__(256) void lstm_init(float* __restrict__ c,
                                                 __bf16* __restrict__ h0) {
    int i = blockIdx.x * blockDim.x + threadIdx.x;
    if (i < BB * HH) {
        c[i] = 0.0f;
        h0[i] = (__bf16)0.0f;
    }
}

// ---------------------------------------------------------------------------
// One LSTM timestep, fully fused:
//   z = [x_t, h_in] @ W + b ; gates ; c, h update.
// 1024 waves; a block's 8 waves tile 4 m_tiles x 2 n_tiles so B addresses
// are shared by 4 waves (WGP$ hits) and A rows by 2 waves.
// Wave (m_tile, n_tile) computes four 16x16 f32 tiles (gates i,j,f,o) with
// v_wmma_f32_16x16x32_bf16, K = 1024 in 32 chunks.
// ---------------------------------------------------------------------------
__global__ __launch_bounds__(256) void lstm_step(
    const float*  __restrict__ x,       // [B, T, D] f32
    const __bf16* __restrict__ WbT,     // [4H, K] bf16 (transposed W)
    const float*  __restrict__ bias,    // [4H]
    const __bf16* __restrict__ hin,     // [B, H] bf16
    __bf16*       __restrict__ hout_bf, // [B, H] bf16 (next step's input)
    float*        __restrict__ c,       // [B, H] f32 cell state
    float*        __restrict__ hout_f,  // [B, H] f32 (d_out)
    int t) {
    const int lane   = threadIdx.x & 31;
    const int wave   = threadIdx.x >> 5;           // 0..7
    // block tiling: 4 m_tiles x 2 n_tiles per block
    const int mi     = wave & 3;                   // 0..3
    const int ni     = wave >> 2;                  // 0..1
    const int mg     = blockIdx.x >> 4;            // 0..7   (m groups of 4)
    const int ng     = blockIdx.x & 15;            // 0..15  (n groups of 2)
    const int m_tile = mg * 4 + mi;                // 0..31
    const int n_tile = ng * 2 + ni;                // 0..31
    const int half   = lane >> 4;                  // 0/1
    const int l16    = lane & 15;
    const int m0     = m_tile * 16;
    const int arow   = m0 + l16;                   // A-matrix row for this lane
    const int col    = n_tile * 16 + l16;          // within-gate output column

    v8f acc0 = {}, acc1 = {}, acc2 = {}, acc3 = {};

    const __bf16* w0 = WbT + (size_t)(0 * HH + col) * KK;  // gate i
    const __bf16* w1 = WbT + (size_t)(1 * HH + col) * KK;  // gate j
    const __bf16* w2 = WbT + (size_t)(2 * HH + col) * KK;  // gate f
    const __bf16* w3 = WbT + (size_t)(3 * HH + col) * KK;  // gate o

    const float*  xrow = x + ((size_t)arow * TT + t) * DD;
    const __bf16* hrow = hin + (size_t)arow * HH;

    // ---- K chunks 0..15 : A comes from x_t (f32 -> bf16 on the fly) ----
#pragma unroll 4
    for (int kc = 0; kc < 16; ++kc) {
        const int kg = kc * 32 + half * 16;
        const float* xp = xrow + kg;
        float4 f0 = *(const float4*)(xp);
        float4 f1 = *(const float4*)(xp + 4);
        float4 f2 = *(const float4*)(xp + 8);
        float4 f3 = *(const float4*)(xp + 12);
        v16bf a;
        a[0]  = (__bf16)f0.x; a[1]  = (__bf16)f0.y; a[2]  = (__bf16)f0.z; a[3]  = (__bf16)f0.w;
        a[4]  = (__bf16)f1.x; a[5]  = (__bf16)f1.y; a[6]  = (__bf16)f1.z; a[7]  = (__bf16)f1.w;
        a[8]  = (__bf16)f2.x; a[9]  = (__bf16)f2.y; a[10] = (__bf16)f2.z; a[11] = (__bf16)f2.w;
        a[12] = (__bf16)f3.x; a[13] = (__bf16)f3.y; a[14] = (__bf16)f3.z; a[15] = (__bf16)f3.w;

        const int kb = kc * 32 + half * 16;
        v16bf b0 = *(const v16bf*)(w0 + kb);
        v16bf b1 = *(const v16bf*)(w1 + kb);
        v16bf b2 = *(const v16bf*)(w2 + kb);
        v16bf b3 = *(const v16bf*)(w3 + kb);
        acc0 = __builtin_amdgcn_wmma_f32_16x16x32_bf16(false, a, false, b0, (short)0, acc0, false, false);
        acc1 = __builtin_amdgcn_wmma_f32_16x16x32_bf16(false, a, false, b1, (short)0, acc1, false, false);
        acc2 = __builtin_amdgcn_wmma_f32_16x16x32_bf16(false, a, false, b2, (short)0, acc2, false, false);
        acc3 = __builtin_amdgcn_wmma_f32_16x16x32_bf16(false, a, false, b3, (short)0, acc3, false, false);
    }

    // ---- K chunks 16..31 : A comes from h_{t-1} (bf16) ----
#pragma unroll 4
    for (int kc = 16; kc < 32; ++kc) {
        const int kg = (kc - 16) * 32 + half * 16;
        v8bf h0v = *(const v8bf*)(hrow + kg);
        v8bf h1v = *(const v8bf*)(hrow + kg + 8);
        v16bf a;
#pragma unroll
        for (int i = 0; i < 8; ++i) { a[i] = h0v[i]; a[8 + i] = h1v[i]; }

        const int kb = kc * 32 + half * 16;
        v16bf b0 = *(const v16bf*)(w0 + kb);
        v16bf b1 = *(const v16bf*)(w1 + kb);
        v16bf b2 = *(const v16bf*)(w2 + kb);
        v16bf b3 = *(const v16bf*)(w3 + kb);
        acc0 = __builtin_amdgcn_wmma_f32_16x16x32_bf16(false, a, false, b0, (short)0, acc0, false, false);
        acc1 = __builtin_amdgcn_wmma_f32_16x16x32_bf16(false, a, false, b1, (short)0, acc1, false, false);
        acc2 = __builtin_amdgcn_wmma_f32_16x16x32_bf16(false, a, false, b2, (short)0, acc2, false, false);
        acc3 = __builtin_amdgcn_wmma_f32_16x16x32_bf16(false, a, false, b3, (short)0, acc3, false, false);
    }

    // ---- fused gate math (gate order: i, j, f, o; forget bias = 1.0) ----
    const float bi = bias[0 * HH + col];
    const float bj = bias[1 * HH + col];
    const float bf = bias[2 * HH + col];
    const float bo = bias[3 * HH + col];
#pragma unroll
    for (int r = 0; r < 8; ++r) {
        const int m = m0 + half * 8 + r;  // C/D layout: VGPR r -> M = r + 8*half
        const size_t idx = (size_t)m * HH + col;
        const float zi = acc0[r] + bi;
        const float zj = acc1[r] + bj;
        const float zf = acc2[r] + bf;
        const float zo = acc3[r] + bo;
        const float co = c[idx];
        const float cn = co * fsig(zf + 1.0f) + fsig(zi) * ftanh(zj);
        const float hn = ftanh(cn) * fsig(zo);
        c[idx]       = cn;
        hout_bf[idx] = (__bf16)hn;
        hout_f[idx]  = hn;   // last step's write == h_T (reference output)
    }
}

extern "C" void kernel_launch(void* const* d_in, const int* in_sizes, int n_in,
                              void* d_out, int out_size, void* d_ws, size_t ws_size,
                              hipStream_t stream) {
    const float* x    = (const float*)d_in[0];  // [512,128,512]
    const float* W    = (const float*)d_in[1];  // [1024,2048]
    const float* bias = (const float*)d_in[2];  // [2048]
    float* out = (float*)d_out;                 // [512,512]

    char* ws = (char*)d_ws;
    __bf16* WbT = (__bf16*)ws;                               // 4 MB
    __bf16* h0  = (__bf16*)(ws + (4u << 20));                // 512 KB
    __bf16* h1  = (__bf16*)(ws + (4u << 20) + (512u << 10)); // 512 KB
    float*  c   = (float*)(ws + (5u << 20));                 // 1 MB

    lstm_convW<<<dim3(KK / 32, NG / 32), dim3(32, 8), 0, stream>>>(W, WbT);
    lstm_init<<<(BB * HH + 255) / 256, 256, 0, stream>>>(c, h0);

    for (int t = 0; t < TT; ++t) {
        const __bf16* hin = (t & 1) ? h1 : h0;
        __bf16*       hob = (t & 1) ? h0 : h1;
        lstm_step<<<dim3(128), dim3(256), 0, stream>>>(x, WbT, bias, hin, hob, c, out, t);
    }
}
